// GeneExpressionEncoder_20693152432938
// MI455X (gfx1250) — matrix operand order repaired and checked by reading, not compiled, not run on hardware
//
#include <hip/hip_runtime.h>

// ---------------- problem constants ----------------
#define GG 33538
#define DD 256
#define OO 512
#define LL 6
#define HH 8
#define KK 2048
#define BB 4
#define FFD 1024
#define TT (KK + 1)          // 2049 tokens
#define MT (BB * TT)         // 8196 rows
#define EPSV 1e-5f
#define QTG 33               // groups of 4 query-tiles: ceil(ceil(2049/16)/4)
#define SSTR 36              // LDS score-row stride (dwords): bank-conflict-free, 16B aligned
#define VSTR 40              // LDS Vt row stride (elements): 80B, 16B-aligned, conflict-free

typedef __bf16 bf16_t;
typedef __attribute__((ext_vector_type(16))) __bf16 v16bf;
typedef __attribute__((ext_vector_type(8)))  float  v8f;

// ---------------- async global->LDS copy tiers ----------------
#if defined(__has_builtin)
#if __has_builtin(__builtin_amdgcn_global_load_async_to_lds_b128)
#define ASYNC_TIER 1
#endif
#endif
#ifndef ASYNC_TIER
#define ASYNC_TIER 2
#endif

typedef int vs4i __attribute__((vector_size(16)));
typedef __attribute__((address_space(1))) vs4i gas_v4i;
typedef __attribute__((address_space(3))) vs4i las_v4i;

__device__ inline void async_cp16(const bf16_t* g, bf16_t* l) {
#if ASYNC_TIER == 1
  __builtin_amdgcn_global_load_async_to_lds_b128((gas_v4i*)g, (las_v4i*)l, 0, 0);
#else
  unsigned loff = (unsigned)(unsigned long long)(las_v4i*)l;
  asm volatile("global_load_async_to_lds_b128 %0, %1, off"
               :: "v"(loff), "v"((unsigned long long)g)
               : "memory");
#endif
}

__device__ inline void wait_async0() {
#if defined(__has_builtin)
#if __has_builtin(__builtin_amdgcn_s_wait_asynccnt)
  __builtin_amdgcn_s_wait_asynccnt(0);
#define HAVE_WAIT_ASYNC 1
#endif
#endif
#ifndef HAVE_WAIT_ASYNC
  asm volatile("s_wait_asynccnt 0" ::: "memory");
#endif
}

// ---------------- small device helpers ----------------
__device__ inline v8f v8f_zero() {
  v8f z;
#pragma unroll
  for (int i = 0; i < 8; i++) z[i] = 0.0f;
  return z;
}

__device__ inline v8f wmma_bf16(v16bf a, v16bf b, v8f c) {
  return __builtin_amdgcn_wmma_f32_16x16x32_bf16(false, a, false, b, (short)0, c, false, false);
}

__device__ inline float gelu_f(float x) {
  return 0.5f * x * (1.0f + erff(x * 0.70710678118654752f));
}

// block (256 threads = 8 waves) sum
__device__ inline float block_sum_256(float v, float* sb) {
#pragma unroll
  for (int m = 16; m >= 1; m >>= 1) v += __shfl_xor(v, m, 32);
  int w = threadIdx.x >> 5;
  if ((threadIdx.x & 31) == 0) sb[w] = v;
  __syncthreads();
  float r = 0.0f;
#pragma unroll
  for (int i = 0; i < 8; i++) r += sb[i];
  __syncthreads();
  return r;
}

__device__ inline unsigned f2key(float f) {
  unsigned u = __float_as_uint(f);
  return (u & 0x80000000u) ? ~u : (u | 0x80000000u);
}

// 1024-thread exclusive scan (Hillis-Steele in LDS); also returns total
__device__ inline unsigned scan1024(unsigned v, unsigned* buf, unsigned& total) {
  int tid = threadIdx.x;
  buf[tid] = v;
  __syncthreads();
  for (int off = 1; off < 1024; off <<= 1) {
    unsigned t = (tid >= off) ? buf[tid - off] : 0u;
    __syncthreads();
    buf[tid] += t;
    __syncthreads();
  }
  unsigned incl = buf[tid];
  total = buf[1023];
  __syncthreads();
  return incl - v;
}

// ---------------- kernel: fp32 -> bf16 convert ----------------
__global__ __launch_bounds__(256) void f32_to_bf16_kernel(const float* __restrict__ src,
                                                          bf16_t* __restrict__ dst, int n) {
  int i = blockIdx.x * 256 + threadIdx.x;
  if (i < n) dst[i] = (bf16_t)src[i];
}

// ---------------- kernel: per-batch top-K selection ----------------
__global__ __launch_bounds__(1024) void topk_kernel(const float* __restrict__ expr,
                                                    int* __restrict__ sel_idx,
                                                    int* __restrict__ counts) {
  __shared__ unsigned hist[256];
  __shared__ unsigned sbuf[1024];
  __shared__ unsigned s_prefix, s_mask, s_want, s_ngt, s_runA, s_runE;
  int b = blockIdx.x;
  int tid = threadIdx.x;
  const float* row = expr + (size_t)b * GG;

  if (tid == 0) { s_prefix = 0u; s_mask = 0u; s_want = KK; s_ngt = 0u; s_runA = 0u; s_runE = 0u; }
  __syncthreads();

  for (int pass = 0; pass < 4; pass++) {
    int shift = 24 - 8 * pass;
    for (int i = tid; i < 256; i += 1024) hist[i] = 0u;
    __syncthreads();
    unsigned pmask = s_mask, pfix = s_prefix;
    for (int g = tid; g < GG; g += 1024) {
      unsigned u = f2key(row[g]);
      if ((u & pmask) == pfix) atomicAdd(&hist[(u >> shift) & 255u], 1u);
    }
    __syncthreads();
    if (tid == 0) {
      unsigned want = s_want, cum = 0u;
      int chosen = 0;
      for (int bin = 255; bin >= 0; bin--) {
        unsigned c = hist[bin];
        if (cum + c >= want) { chosen = bin; break; }
        cum += c;
      }
      s_ngt += cum;
      s_want = want - cum;
      s_prefix |= ((unsigned)chosen) << shift;
      s_mask |= 0xFFu << shift;
    }
    __syncthreads();
  }

  unsigned kth = s_prefix;
  unsigned tieneed = (unsigned)KK - s_ngt;

  for (int base = 0; base < GG; base += 1024) {
    int g = base + tid;
    bool inb = g < GG;
    float v = inb ? row[g] : 0.0f;
    unsigned u = inb ? f2key(v) : 0u;
    unsigned eq = (inb && u == kth) ? 1u : 0u;
    unsigned gt = (inb && u > kth) ? 1u : 0u;
    unsigned teq;
    unsigned excl_eq = scan1024(eq, sbuf, teq);
    unsigned member = gt | (eq & (((s_runE + excl_eq) < tieneed) ? 1u : 0u));
    unsigned act = (member && (v > 0.0f)) ? 1u : 0u;
    unsigned tact;
    unsigned excl_a = scan1024(act, sbuf, tact);
    if (act) sel_idx[(size_t)b * KK + s_runA + excl_a] = g;
    __syncthreads();
    if (tid == 0) { s_runA += tact; s_runE += teq; }
    __syncthreads();
  }
  if (tid == 0) counts[b] = (int)s_runA;
}

// ---------------- kernel: build token embeddings (fp32) ----------------
__global__ __launch_bounds__(256) void embed_kernel(
    const float* __restrict__ expr, const float* __restrict__ gene_emb,
    const float* __restrict__ ep_w1, const float* __restrict__ ep_b1,
    const float* __restrict__ ep_w2, const float* __restrict__ ep_b2,
    const float* __restrict__ cls, const int* __restrict__ sel_idx,
    const int* __restrict__ counts, float* __restrict__ X) {
  int token = blockIdx.x % TT;
  int b = blockIdx.x / TT;
  int d = threadIdx.x;
  float* xrow = X + ((size_t)b * TT + token) * DD;
  if (token == 0) { xrow[d] = cls[d]; return; }
  int j = token - 1;
  if (j >= counts[b]) { xrow[d] = 0.0f; return; }
  int g = sel_idx[(size_t)b * KK + j];
  float e = expr[(size_t)b * GG + g];
  __shared__ float h[DD];
  h[d] = gelu_f(e * ep_w1[d] + ep_b1[d]);
  __syncthreads();
  float acc = 0.0f;
  const float* wrow = ep_w2 + (size_t)d * DD;
#pragma unroll 4
  for (int dd = 0; dd < DD; dd++) acc += h[dd] * wrow[dd];
  xrow[d] = gene_emb[(size_t)g * DD + d] + acc + ep_b2[d];
}

// ---------------- kernel: LayerNorm (fp32 in -> bf16 out) ----------------
__global__ __launch_bounds__(256) void ln_kernel(const float* __restrict__ X,
                                                 const float* __restrict__ s,
                                                 const float* __restrict__ bp,
                                                 bf16_t* __restrict__ Y) {
  __shared__ float sb[8];
  size_t row = blockIdx.x;
  int d = threadIdx.x;
  float x = X[row * DD + d];
  float mean = block_sum_256(x, sb) * (1.0f / DD);
  float c = x - mean;
  float var = block_sum_256(c * c, sb) * (1.0f / DD);
  float y = c * rsqrtf(var + EPSV) * s[d] + bp[d];
  Y[row * DD + d] = (bf16_t)y;
}

// ---------------- kernel: bf16 WMMA GEMM, C = A(M,K) * W(N,K)^T + bias ----------------
// EPI 0: store bf16; EPI 1: residual add into fp32 X; EPI 2: gelu -> bf16
template <int EPI>
__global__ __launch_bounds__(256) void gemm_bf16_kernel(
    const bf16_t* __restrict__ A, const bf16_t* __restrict__ W,
    const float* __restrict__ bias, bf16_t* __restrict__ Cb,
    float* __restrict__ Xres, int Mtot, int N, int Kdim, int waves_total) {
  int gw = blockIdx.x * 8 + (threadIdx.x >> 5);
  if (gw >= waves_total) return;
  int lane = threadIdx.x & 31;
  int ntiles = N >> 6;
  int mt = gw / ntiles, nt = gw - mt * ntiles;
  int m0 = mt * 16, n0 = nt * 64;
  int ml = lane & 15, hg = lane >> 4;
  int mrow = m0 + ml;
  bool mv = (mrow < Mtot);
  const bf16_t* arow = A + (size_t)mrow * Kdim;
  int ncol = n0 + ml;

  v8f acc[4];
#pragma unroll
  for (int j = 0; j < 4; j++) acc[j] = v8f_zero();

  for (int kc = 0; kc < Kdim; kc += 32) {
    v16bf a;
    if (mv) {
#pragma unroll
      for (int e = 0; e < 8; e++) a[e] = arow[kc + hg * 8 + e];
#pragma unroll
      for (int e = 0; e < 8; e++) a[8 + e] = arow[kc + 16 + hg * 8 + e];
    } else {
#pragma unroll
      for (int e = 0; e < 16; e++) a[e] = (bf16_t)0.0f;
    }
    if (kc + 32 < Kdim)
      __builtin_prefetch(W + (size_t)ncol * Kdim + kc + 32 + hg * 16, 0, 1);
#pragma unroll
    for (int j = 0; j < 4; j++) {
      const bf16_t* wrow = W + (size_t)(ncol + j * 16) * Kdim + kc;
      v16bf bm;
#pragma unroll
      for (int e = 0; e < 16; e++) bm[e] = wrow[(e < 8) ? (hg * 8 + e) : (16 + hg * 8 + e - 8)];
      acc[j] = wmma_bf16(a, bm, acc[j]);
    }
  }

#pragma unroll
  for (int j = 0; j < 4; j++) {
    int n = ncol + j * 16;
    float bs = bias ? bias[n] : 0.0f;
#pragma unroll
    for (int r = 0; r < 8; r++) {
      int m = m0 + r + 8 * hg;
      if (m < Mtot) {
        float v = acc[j][r] + bs;
        if (EPI == 0) Cb[(size_t)m * N + n] = (bf16_t)v;
        else if (EPI == 1) Xres[(size_t)m * N + n] += v;
        else Cb[(size_t)m * N + n] = (bf16_t)gelu_f(v);
      }
    }
  }
}

// ---------------- kernel: flash attention ----------------
// One 128-thread block = 4 waves = 4 consecutive 16-query tiles of one (b,h).
// K chunk (32 keys x 32 dh, bf16) staged into LDS via async global->LDS copy;
// V chunk staged TRANSPOSED (Vt[dh][key], stride-40 rows) so the P x V
// B-operand loads are contiguous ds_load_b128. Softmax is row-per-lane.
__global__ __launch_bounds__(128) void attn_kernel(const bf16_t* __restrict__ QKV,
                                                   bf16_t* __restrict__ O,
                                                   const int* __restrict__ counts) {
  __shared__ bf16_t Klds[32 * 32];
  __shared__ bf16_t Vt[32 * VSTR];  // transposed: row = dh (0..31), col = key (0..31)
  __shared__ float Sl[4][16 * SSTR];

  int grp = blockIdx.x % QTG;
  int tmp = blockIdx.x / QTG;
  int h = tmp % HH;
  int b = tmp / HH;
  int wv = threadIdx.x >> 5;
  int lane = threadIdx.x & 31;
  float* Sw = Sl[wv];
  int t0 = grp * 64 + wv * 16;
  int vl = counts[b] + 1;  // valid tokens (cls + actives)
  const bf16_t* base = QKV + (size_t)b * TT * (3 * DD);
  int ml = lane & 15, hg = lane >> 4;

  // load Q tile (A operand), rows = queries, K-dim = dh=32
  int tq = t0 + ml;
  v16bf qa;
  if (tq < TT) {
    const bf16_t* qrow = base + (size_t)tq * (3 * DD) + h * 32;
#pragma unroll
    for (int e = 0; e < 8; e++) qa[e] = qrow[hg * 8 + e];
#pragma unroll
    for (int e = 0; e < 8; e++) qa[8 + e] = qrow[16 + hg * 8 + e];
  } else {
#pragma unroll
    for (int e = 0; e < 16; e++) qa[e] = (bf16_t)0.0f;
  }

  // staging assignment: thread -> (key row 0..31, 8-bf16 quarter 0..3)
  int srow = threadIdx.x >> 2;
  int squad = threadIdx.x & 3;
  bf16_t* kdst = Klds + srow * 32 + squad * 8;

  v8f o0 = v8f_zero(), o1 = v8f_zero();
  float rmloc = -3.0e38f, rlloc = 0.0f;  // softmax stats for row ml (this lane)
  const float scale = 0.17677669529663687f;  // 1/sqrt(32)

  for (int kt = 0; kt < vl; kt += 32) {
    // ---- cooperative staging of K (async) and transposed V ----
    int trow = kt + srow;
    if (trow >= TT) trow = TT - 1;  // clamp: finite data; masked by score bias
    const bf16_t* grow = base + (size_t)trow * (3 * DD) + h * 32 + squad * 8;
    async_cp16(grow + DD, kdst);
    {
      const uint4 vv = *(const uint4*)(grow + 2 * DD);
      bf16_t vtmp[8];
      *(uint4*)vtmp = vv;
#pragma unroll
      for (int i = 0; i < 8; i++) Vt[(squad * 8 + i) * VSTR + srow] = vtmp[i];
    }
    wait_async0();
    __syncthreads();

    // ---- scores: B operand from Klds, column n = key ----
    v16bf kb0, kb1;
#pragma unroll
    for (int e = 0; e < 16; e++) kb0[e] = Klds[ml * 32 + ((e < 8) ? (hg * 8 + e) : (16 + hg * 8 + e - 8))];
#pragma unroll
    for (int e = 0; e < 16; e++) kb1[e] = Klds[(16 + ml) * 32 + ((e < 8) ? (hg * 8 + e) : (16 + hg * 8 + e - 8))];

    v8f s0 = wmma_bf16(qa, kb0, v8f_zero());
    v8f s1 = wmma_bf16(qa, kb1, v8f_zero());
    int nk0 = kt + ml, nk1 = kt + 16 + ml;
    float bias0 = (nk0 < vl) ? 0.0f : -1.0e9f;
    float bias1 = (nk1 < vl) ? 0.0f : -1.0e9f;

    // scaled+masked scores -> LDS in C layout (row stride SSTR, conflict-free)
#pragma unroll
    for (int r = 0; r < 8; r++) {
      Sw[(r + 8 * hg) * SSTR + ml] = s0[r] * scale + bias0;
      Sw[(r + 8 * hg) * SSTR + 16 + ml] = s1[r] * scale + bias1;
    }
    asm volatile("s_wait_dscnt 0" ::: "memory");

    // row-per-lane softmax over this chunk
    float rv[32];
#pragma unroll
    for (int j = 0; j < 32; j++) rv[j] = Sw[ml * SSTR + j];
    float mx = rv[0];
#pragma unroll
    for (int j = 1; j < 32; j++) mx = fmaxf(mx, rv[j]);
    float mnew = fmaxf(rmloc, mx);
    float alpha = __expf(rmloc - mnew);
    rmloc = mnew;
    float ps = 0.0f;
#pragma unroll
    for (int j = 0; j < 32; j++) { rv[j] = __expf(rv[j] - mnew); ps += rv[j]; }
    rlloc = rlloc * alpha + ps;

    // build P A-operand directly from this lane's row
    v16bf pa;
#pragma unroll
    for (int e = 0; e < 8; e++) {
      pa[e] = (bf16_t)rv[hg * 8 + e];
      pa[8 + e] = (bf16_t)rv[16 + hg * 8 + e];
    }

    // rescale accumulators: alpha for row (r+8*hg) lives in lane (r+8*hg)
#pragma unroll
    for (int r = 0; r < 8; r++) {
      float ar = __shfl(alpha, 8 * hg + r, 32);
      o0[r] = o0[r] * ar;
      o1[r] = o1[r] * ar;
    }

    // V tile as B operands from Vt (contiguous rows: n = dh, k = key)
    v16bf vb0, vb1;
#pragma unroll
    for (int e = 0; e < 16; e++) vb0[e] = Vt[ml * VSTR + ((e < 8) ? (hg * 8 + e) : (16 + hg * 8 + e - 8))];
#pragma unroll
    for (int e = 0; e < 16; e++) vb1[e] = Vt[(16 + ml) * VSTR + ((e < 8) ? (hg * 8 + e) : (16 + hg * 8 + e - 8))];
    o0 = wmma_bf16(pa, vb0, o0);
    o1 = wmma_bf16(pa, vb1, o1);

    __syncthreads();  // protect Klds/Vt before next staging
  }

#pragma unroll
  for (int r = 0; r < 8; r++) {
    int t = t0 + r + 8 * hg;
    if (t < TT) {
      float inv = 1.0f / __shfl(rlloc, 8 * hg + r, 32);
      bf16_t* orow = O + ((size_t)b * TT + t) * DD + h * 32;
      orow[ml] = (bf16_t)(o0[r] * inv);
      orow[16 + ml] = (bf16_t)(o1[r] * inv);
    }
  }
}

// ---------------- kernel: final LN + projection head (fp32, M=4) ----------------
__global__ __launch_bounds__(256) void head_kernel(const float* __restrict__ X,
                                                   const float* __restrict__ fn_s,
                                                   const float* __restrict__ fn_b,
                                                   const float* __restrict__ w1,
                                                   const float* __restrict__ b1,
                                                   const float* __restrict__ w2,
                                                   const float* __restrict__ b2,
                                                   float* __restrict__ out) {
  __shared__ float sb[8];
  __shared__ float xr[DD];
  __shared__ float hid[OO];
  int b = blockIdx.x;
  int tid = threadIdx.x;
  float x = X[(size_t)b * TT * DD + tid];
  float mean = block_sum_256(x, sb) * (1.0f / DD);
  float c = x - mean;
  float var = block_sum_256(c * c, sb) * (1.0f / DD);
  xr[tid] = c * rsqrtf(var + EPSV) * fn_s[tid] + fn_b[tid];
  __syncthreads();
  for (int o = tid; o < OO; o += 256) {
    float a = 0.0f;
    const float* wr = w1 + (size_t)o * DD;
#pragma unroll 4
    for (int d = 0; d < DD; d++) a += xr[d] * wr[d];
    hid[o] = gelu_f(a + b1[o]);
  }
  __syncthreads();
  for (int o = tid; o < OO; o += 256) {
    float a = 0.0f;
    const float* wr = w2 + (size_t)o * OO;
#pragma unroll 4
    for (int d = 0; d < OO; d++) a += hid[d] * wr[d];
    out[(size_t)b * OO + o] = a + b2[o];
  }
}

// ---------------- host launcher ----------------
extern "C" void kernel_launch(void* const* d_in, const int* in_sizes, int n_in,
                              void* d_out, int out_size, void* d_ws, size_t ws_size,
                              hipStream_t stream) {
  const float* expr    = (const float*)d_in[0];
  const float* gene_emb= (const float*)d_in[1];
  const float* ep_w1   = (const float*)d_in[2];
  const float* ep_b1   = (const float*)d_in[3];
  const float* ep_w2   = (const float*)d_in[4];
  const float* ep_b2   = (const float*)d_in[5];
  const float* cls     = (const float*)d_in[6];
  const float* ln1_s   = (const float*)d_in[7];
  const float* ln1_b   = (const float*)d_in[8];
  const float* qkv_w   = (const float*)d_in[9];
  const float* qkv_b   = (const float*)d_in[10];
  const float* out_w   = (const float*)d_in[11];
  const float* out_b   = (const float*)d_in[12];
  const float* ln2_s   = (const float*)d_in[13];
  const float* ln2_b   = (const float*)d_in[14];
  const float* ff_w1   = (const float*)d_in[15];
  const float* ff_b1   = (const float*)d_in[16];
  const float* ff_w2   = (const float*)d_in[17];
  const float* ff_b2   = (const float*)d_in[18];
  const float* fn_s    = (const float*)d_in[19];
  const float* fn_b    = (const float*)d_in[20];
  const float* pr_w1   = (const float*)d_in[21];
  const float* pr_b1   = (const float*)d_in[22];
  const float* pr_w2   = (const float*)d_in[23];
  const float* pr_b2   = (const float*)d_in[24];
  float* out = (float*)d_out;

  char* ws = (char*)d_ws;
  size_t off = 0;
  auto alloc = [&](size_t bytes) -> char* {
    off = (off + 255) & ~(size_t)255;
    char* p = ws + off;
    off += bytes;
    return p;
  };

  int*    sel   = (int*)alloc((size_t)BB * KK * sizeof(int));
  int*    cnt   = (int*)alloc(64);
  float*  X     = (float*)alloc((size_t)MT * DD * sizeof(float));
  bf16_t* Ybf   = (bf16_t*)alloc((size_t)MT * DD * sizeof(bf16_t));     // LN out / attn out
  bf16_t* QKVbf = (bf16_t*)alloc((size_t)MT * 3 * DD * sizeof(bf16_t));
  bf16_t* Hbf   = (bf16_t*)alloc((size_t)MT * FFD * sizeof(bf16_t));
  bf16_t* Wqkv  = (bf16_t*)alloc((size_t)LL * 3 * DD * DD * sizeof(bf16_t));
  bf16_t* Wout  = (bf16_t*)alloc((size_t)LL * DD * DD * sizeof(bf16_t));
  bf16_t* Wf1   = (bf16_t*)alloc((size_t)LL * FFD * DD * sizeof(bf16_t));
  bf16_t* Wf2   = (bf16_t*)alloc((size_t)LL * DD * FFD * sizeof(bf16_t));

  auto conv = [&](const float* s, bf16_t* dvec, size_t n) {
    int blocks = (int)((n + 255) / 256);
    f32_to_bf16_kernel<<<blocks, 256, 0, stream>>>(s, dvec, (int)n);
  };
  conv(qkv_w, Wqkv, (size_t)LL * 3 * DD * DD);
  conv(out_w, Wout, (size_t)LL * DD * DD);
  conv(ff_w1, Wf1, (size_t)LL * FFD * DD);
  conv(ff_w2, Wf2, (size_t)LL * DD * FFD);

  topk_kernel<<<BB, 1024, 0, stream>>>(expr, sel, cnt);
  embed_kernel<<<BB * TT, 256, 0, stream>>>(expr, gene_emb, ep_w1, ep_b1, ep_w2, ep_b2,
                                            cls, sel, cnt, X);

  const int mtiles = (MT + 15) / 16;
  auto gemm = [&](int epi, const bf16_t* A, const bf16_t* Wm, const float* bias,
                  bf16_t* Cb, float* Xr, int N, int Kd) {
    int waves = mtiles * (N / 64);
    int blocks = (waves + 7) / 8;
    if (epi == 0)
      gemm_bf16_kernel<0><<<blocks, 256, 0, stream>>>(A, Wm, bias, Cb, Xr, MT, N, Kd, waves);
    else if (epi == 1)
      gemm_bf16_kernel<1><<<blocks, 256, 0, stream>>>(A, Wm, bias, Cb, Xr, MT, N, Kd, waves);
    else
      gemm_bf16_kernel<2><<<blocks, 256, 0, stream>>>(A, Wm, bias, Cb, Xr, MT, N, Kd, waves);
  };

  const int attn_blocks = BB * HH * QTG;

  for (int l = 0; l < LL; l++) {
    ln_kernel<<<MT, 256, 0, stream>>>(X, ln1_s + (size_t)l * DD, ln1_b + (size_t)l * DD, Ybf);
    gemm(0, Ybf, Wqkv + (size_t)l * 3 * DD * DD, qkv_b + (size_t)l * 3 * DD, QKVbf, nullptr,
         3 * DD, DD);
    attn_kernel<<<attn_blocks, 128, 0, stream>>>(QKVbf, Ybf, cnt);
    gemm(1, Ybf, Wout + (size_t)l * DD * DD, out_b + (size_t)l * DD, nullptr, X, DD, DD);
    ln_kernel<<<MT, 256, 0, stream>>>(X, ln2_s + (size_t)l * DD, ln2_b + (size_t)l * DD, Ybf);
    gemm(2, Ybf, Wf1 + (size_t)l * FFD * DD, ff_b1 + (size_t)l * FFD, Hbf, nullptr, FFD, DD);
    gemm(1, Hbf, Wf2 + (size_t)l * DD * FFD, ff_b2 + (size_t)l * DD, nullptr, X, DD, FFD);
  }

  head_kernel<<<BB, 256, 0, stream>>>(X, fn_s, fn_b, pr_w1, pr_b1, pr_w2, pr_b2, out);
}
